// TransformerBlock_6605659701472
// MI455X (gfx1250) — compile-verified, matrix operand order checked by reading, not run on hardware
//
#include <hip/hip_runtime.h>
#include <hip/hip_bf16.h>

// ---------------------------------------------------------------------------
// Types for WMMA fragments (gfx1250, wave32)
// ---------------------------------------------------------------------------
typedef _Float16 h8  __attribute__((ext_vector_type(8)));
typedef _Float16 h16 __attribute__((ext_vector_type(16)));
typedef float    f8  __attribute__((ext_vector_type(8)));
typedef int      v4i __attribute__((ext_vector_type(4)));

union H16U { h16 v; h8 h[2]; };

// Load a 16x32 f16 fragment (A-layout == B-layout for [rows][K] row-major src).
// lanes 0-15: row = lane,  kb = 0 ; lanes 16-31: row = lane-16, kb = 8
__device__ __forceinline__ h16 load_frag16x32(const _Float16* __restrict__ p,
                                              int ld, int lane) {
  const int m  = lane & 15;
  const int kb = (lane < 16) ? 0 : 8;
  H16U r;
  r.h[0] = *(const h8*)(p + (size_t)m * ld + kb);
  r.h[1] = *(const h8*)(p + (size_t)m * ld + 16 + kb);
  return r.v;
}

__device__ __forceinline__ f8 wmma_f16(h16 a, h16 b, f8 c) {
  return __builtin_amdgcn_wmma_f32_16x16x32_f16(false, a, false, b,
                                                (short)0, c, false, false);
}

// ---------------------------------------------------------------------------
// Async global->LDS staging (CDNA5 ASYNCcnt path), guarded so compilation
// falls back to synchronous staging if the builtins are unavailable.
// Builtin signature (from hipcc diagnostics): (int4 AS1*, int4 AS3*, imm, imm)
// ---------------------------------------------------------------------------
#if defined(__gfx1250__) && \
    __has_builtin(__builtin_amdgcn_global_load_async_to_lds_b128) && \
    __has_builtin(__builtin_amdgcn_s_wait_asynccnt)
#define ASYNC_LDS 1
#else
#define ASYNC_LDS 0
#endif

__device__ __forceinline__ void cp16_to_lds(const _Float16* g, _Float16* l) {
#if ASYNC_LDS
  __builtin_amdgcn_global_load_async_to_lds_b128(
      (__attribute__((address_space(1))) v4i*)g,
      (__attribute__((address_space(3))) v4i*)l, 0, 0);
#else
  *(h8*)l = *(const h8*)g;
#endif
}

template <int N> __device__ __forceinline__ void async_wait() {
#if ASYNC_LDS
  __builtin_amdgcn_s_wait_asynccnt(N);
#endif
}

// ---------------------------------------------------------------------------
// Problem constants
// ---------------------------------------------------------------------------
#define BB   2
#define SS   2048
#define EE   1024
#define HH   16
#define DH   64
#define MROWS (BB * SS)               // 4096
#define QKV_ONE (BB * HH * SS * DH)   // 4194304 elements per Q/K/V tensor

// ---------------------------------------------------------------------------
// LayerNorm: fp32 in -> f16 out, one block (256 thr) per row of 1024
// ---------------------------------------------------------------------------
__global__ __launch_bounds__(256) void ln_kernel(
    const float* __restrict__ x, const float* __restrict__ g,
    const float* __restrict__ b, _Float16* __restrict__ out) {
  __shared__ float red[256];
  const int row = blockIdx.x;
  const int tid = threadIdx.x;
  const float* xp = x + (size_t)row * EE;

  float local[4];
  float s = 0.f;
#pragma unroll
  for (int i = 0; i < 4; ++i) { local[i] = xp[tid + i * 256]; s += local[i]; }
  red[tid] = s; __syncthreads();
  for (int off = 128; off > 0; off >>= 1) {
    if (tid < off) red[tid] += red[tid + off];
    __syncthreads();
  }
  const float mu = red[0] * (1.f / (float)EE);
  __syncthreads();

  float v = 0.f;
#pragma unroll
  for (int i = 0; i < 4; ++i) { float d = local[i] - mu; v += d * d; }
  red[tid] = v; __syncthreads();
  for (int off = 128; off > 0; off >>= 1) {
    if (tid < off) red[tid] += red[tid + off];
    __syncthreads();
  }
  const float rstd = rsqrtf(red[0] * (1.f / (float)EE) + 1e-5f);

#pragma unroll
  for (int i = 0; i < 4; ++i) {
    int c = tid + i * 256;
    out[(size_t)row * EE + c] =
        (_Float16)((local[i] - mu) * rstd * g[c] + b[c]);
  }
}

// ---------------------------------------------------------------------------
// Weight prep: fp32 [K,N] -> f16 transposed [N,K]
// ---------------------------------------------------------------------------
__global__ void transpose_f16(const float* __restrict__ in,
                              _Float16* __restrict__ out, int N, int K) {
  size_t idx = (size_t)blockIdx.x * 256 + threadIdx.x;
  if (idx >= (size_t)N * K) return;
  size_t n = idx / K, k = idx % K;
  out[idx] = (_Float16)in[k * (size_t)N + n];
}

// Wq/Wk/Wv [H,E,DH] fp32 -> fused transposed f16 [3*H*DH, E]
__global__ void convert_qkv_w(const float* __restrict__ Wq,
                              const float* __restrict__ Wk,
                              const float* __restrict__ Wv,
                              _Float16* __restrict__ out) {
  size_t idx = (size_t)blockIdx.x * 256 + threadIdx.x;
  if (idx >= (size_t)3 * EE * EE) return;
  int e = (int)(idx & (EE - 1));
  int n = (int)(idx >> 10);          // 0..3071
  int w = n >> 10;
  int hd = n & 1023;
  int h = hd >> 6, d = hd & 63;
  const float* W = (w == 0) ? Wq : ((w == 1) ? Wk : Wv);
  out[idx] = (_Float16)W[((size_t)h * EE + e) * DH + d];
}

// ---------------------------------------------------------------------------
// WMMA GEMM: C[M,N] = A[M,K](f16,row) x Bt[N,K](f16,row) with epilogue
//   EPI 0: scatter to Q [B,H,S,DH], K [B,H,S,DH], Vt [B,H,DH,S]   (f16)
//   EPI 1: out_f32[row,col] = v + bias[col] + resid[row,col]      (f32)
//   EPI 2: out_f16[row,col] = relu(v + bias[col])                 (f16)
// Block: 256 thr = 8 waves (4 x 2), tile 128x64, K-step 32.
// Double-buffered LDS; tiles staged with GLOBAL_LOAD_ASYNC_TO_LDS_B128 and
// synchronized with s_wait_asynccnt when available.
// ---------------------------------------------------------------------------
#define BM 128
#define BN 64
#define BK 32
#define LDP (BK + 8)   // padded LDS row (halfs), 80B => 16B aligned chunks

template <int EPI>
__global__ __launch_bounds__(256) void wmma_gemm(
    const _Float16* __restrict__ A, const _Float16* __restrict__ Bt,
    const float* __restrict__ bias, const float* __restrict__ resid,
    void* __restrict__ outp, int M, int N, int K) {
  __shared__ _Float16 As[2][BM][LDP];
  __shared__ _Float16 Bs[2][BN][LDP];

  const int tid  = threadIdx.x;
  const int lane = tid & 31;
  const int wid  = tid >> 5;
  const int wm   = wid & 3;           // 0..3  (M direction)
  const int wn   = wid >> 2;          // 0..1  (N direction)
  const int m0   = blockIdx.x * BM;
  const int n0   = blockIdx.y * BN;

  // per-thread staging assignment: 3 x 16B chunks per tile
  const int ar = tid >> 2;            // 0..63
  const int ac = (tid & 3) * 8;       // halfs offset 0/8/16/24

  auto issue_tile = [&](int k0, int buf) {
    cp16_to_lds(&A[(size_t)(m0 + ar) * K + k0 + ac], &As[buf][ar][ac]);
    cp16_to_lds(&A[(size_t)(m0 + 64 + ar) * K + k0 + ac],
                &As[buf][64 + ar][ac]);
    cp16_to_lds(&Bt[(size_t)(n0 + ar) * K + k0 + ac], &Bs[buf][ar][ac]);
  };

  f8 acc[2][2] = {};

  issue_tile(0, 0);
  int cb = 0;
  for (int k0 = 0; k0 < K; k0 += BK) {
    if (k0 + BK < K) {
      issue_tile(k0 + BK, cb ^ 1);   // prefetch next tile into other buffer
      async_wait<3>();               // current tile's 3 loads have landed
    } else {
      async_wait<0>();
    }
    __syncthreads();

    h16 af[2], bf[2];
#pragma unroll
    for (int i = 0; i < 2; ++i) {
      af[i] = load_frag16x32(&As[cb][wm * 32 + i * 16][0], LDP, lane);
      bf[i] = load_frag16x32(&Bs[cb][wn * 32 + i * 16][0], LDP, lane);
    }
#pragma unroll
    for (int i = 0; i < 2; ++i)
#pragma unroll
      for (int j = 0; j < 2; ++j)
        acc[i][j] = wmma_f16(af[i], bf[j], acc[i][j]);
    __syncthreads();                 // all waves done reading buf[cb]
    cb ^= 1;
  }

  // epilogue (C/D layout: lane col = lane&15, rows = rbase..rbase+7)
  const int rbase = (lane < 16) ? 0 : 8;
  const int cn    = lane & 15;
#pragma unroll
  for (int i = 0; i < 2; ++i)
#pragma unroll
    for (int j = 0; j < 2; ++j)
#pragma unroll
      for (int r = 0; r < 8; ++r) {
        const int row = m0 + wm * 32 + i * 16 + rbase + r;
        const int col = n0 + wn * 32 + j * 16 + cn;
        const float v = acc[i][j][r];
        if (EPI == 0) {
          _Float16* qkv = (_Float16*)outp;
          int which = col >> 10;
          int hd = col & 1023, h = hd >> 6, d = hd & 63;
          int b = row >> 11, s = row & (SS - 1);
          size_t idx;
          if (which < 2)
            idx = (((size_t)b * HH + h) * SS + s) * DH + d;
          else
            idx = (((size_t)b * HH + h) * DH + d) * SS + s;
          qkv[(size_t)which * QKV_ONE + idx] = (_Float16)v;
        } else if (EPI == 1) {
          float* o = (float*)outp;
          o[(size_t)row * N + col] =
              v + bias[col] + resid[(size_t)row * N + col];
        } else {
          _Float16* o = (_Float16*)outp;
          float t = v + bias[col];
          o[(size_t)row * N + col] = (_Float16)(t > 0.f ? t : 0.f);
        }
      }
}

// ---------------------------------------------------------------------------
// Flash attention, causal. Grid: (S/64, B*H), block = 128 thr = 4 waves.
// Each wave owns 16 query rows; K-tiles of 32 with online softmax.
// Q,K: [B,H,S,DH] f16.  Vt: [B,H,DH,S] f16.  AO: [B,S,H*DH] f16.
// ---------------------------------------------------------------------------
__global__ __launch_bounds__(128) void attn_kernel(
    const _Float16* __restrict__ Q, const _Float16* __restrict__ Kt,
    const _Float16* __restrict__ Vt, _Float16* __restrict__ AO) {
  __shared__ _Float16 pbuf[4][16][40];   // per-wave P bounce (padded)

  const int lane  = threadIdx.x & 31;
  const int w     = threadIdx.x >> 5;
  const int bh    = blockIdx.y;                 // 0..B*H-1
  const int b     = bh >> 4, h = bh & 15;
  const int qbase = blockIdx.x * 64 + w * 16;

  const _Float16* qp = Q  + ((size_t)bh * SS + qbase) * DH;
  const _Float16* kp = Kt + (size_t)bh * SS * DH;
  const _Float16* vp = Vt + (size_t)bh * DH * SS;

  // Q fragments covering K-dim 0..63
  h16 aq0 = load_frag16x32(qp, DH, lane);
  h16 aq1 = load_frag16x32(qp + 32, DH, lane);

  float mrow[8], lrow[8];
  f8 acc[4] = {};
#pragma unroll
  for (int r = 0; r < 8; ++r) { mrow[r] = -__builtin_inff(); lrow[r] = 0.f; }

  const int rbase = (lane < 16) ? 0 : 8;
  const int cn    = lane & 15;
  const float scale = 0.03125f;   // E^-0.5 = 1024^-0.5 (reference scaling)

  for (int t0 = 0; t0 <= qbase + 15; t0 += 32) {
    // ---- scores: QK^T for 16 rows x 32 keys ----
    f8 s[2] = {};
#pragma unroll
    for (int j = 0; j < 2; ++j) {
      h16 bk0 = load_frag16x32(kp + (size_t)(t0 + j * 16) * DH, DH, lane);
      h16 bk1 = load_frag16x32(kp + (size_t)(t0 + j * 16) * DH + 32, DH, lane);
      s[j] = wmma_f16(aq0, bk0, s[j]);
      s[j] = wmma_f16(aq1, bk1, s[j]);
    }
    // scale + causal mask
#pragma unroll
    for (int j = 0; j < 2; ++j)
#pragma unroll
      for (int r = 0; r < 8; ++r) {
        int row = qbase + rbase + r;
        int col = t0 + j * 16 + cn;
        float v = s[j][r] * scale;
        s[j][r] = (col <= row) ? v : -__builtin_inff();
      }
    // ---- online softmax update (row reductions within 16-lane half) ----
#pragma unroll
    for (int r = 0; r < 8; ++r) {
      float tm = fmaxf(s[0][r], s[1][r]);
#pragma unroll
      for (int off = 1; off < 16; off <<= 1)
        tm = fmaxf(tm, __shfl_xor(tm, off, 32));
      float mnew = fmaxf(mrow[r], tm);
      float f    = __expf(mrow[r] - mnew);
      float p0   = __expf(s[0][r] - mnew);
      float p1   = __expf(s[1][r] - mnew);
      s[0][r] = p0; s[1][r] = p1;
      float ps = p0 + p1;
#pragma unroll
      for (int off = 1; off < 16; off <<= 1)
        ps += __shfl_xor(ps, off, 32);
      lrow[r] = lrow[r] * f + ps;
      mrow[r] = mnew;
#pragma unroll
      for (int n = 0; n < 4; ++n) acc[n][r] *= f;
    }
    // ---- P: C/D-layout f32 -> A-layout f16 via wave-private LDS ----
#pragma unroll
    for (int j = 0; j < 2; ++j)
#pragma unroll
      for (int r = 0; r < 8; ++r)
        pbuf[w][rbase + r][j * 16 + cn] = (_Float16)s[j][r];
    asm volatile("s_wait_dscnt 0x0" ::: "memory");   // same-wave LDS RAW
    h16 pa = load_frag16x32(&pbuf[w][0][0], 40, lane);
    // ---- PV: accumulate O over 64 head dims ----
#pragma unroll
    for (int n = 0; n < 4; ++n) {
      h16 bv = load_frag16x32(vp + (size_t)(n * 16) * SS + t0, SS, lane);
      acc[n] = wmma_f16(pa, bv, acc[n]);
    }
  }

  // ---- normalize + store concat heads: AO[b][s][h*64 + d] ----
  _Float16* aop = AO + ((size_t)b * SS + qbase) * (HH * DH) + h * DH;
#pragma unroll
  for (int n = 0; n < 4; ++n)
#pragma unroll
    for (int r = 0; r < 8; ++r) {
      float v = acc[n][r] / lrow[r];
      aop[(size_t)(rbase + r) * (HH * DH) + n * 16 + cn] = (_Float16)v;
    }
}

// ---------------------------------------------------------------------------
// Host-side orchestration
// ---------------------------------------------------------------------------
extern "C" void kernel_launch(void* const* d_in, const int* in_sizes, int n_in,
                              void* d_out, int out_size, void* d_ws,
                              size_t ws_size, hipStream_t stream) {
  (void)in_sizes; (void)n_in; (void)out_size; (void)ws_size;
  const float* x     = (const float*)d_in[0];
  const float* ln1_g = (const float*)d_in[1];
  const float* ln1_b = (const float*)d_in[2];
  const float* Wq    = (const float*)d_in[3];
  const float* Wk    = (const float*)d_in[4];
  const float* Wv    = (const float*)d_in[5];
  const float* Wo    = (const float*)d_in[6];
  const float* bo    = (const float*)d_in[7];
  const float* ln2_g = (const float*)d_in[8];
  const float* ln2_b = (const float*)d_in[9];
  const float* W1    = (const float*)d_in[10];
  const float* b1    = (const float*)d_in[11];
  const float* W2    = (const float*)d_in[12];
  const float* b2    = (const float*)d_in[13];
  float* out = (float*)d_out;

  char* ws = (char*)d_ws;
  auto alloc = [&](size_t bytes) {
    char* p = ws;
    ws += (bytes + 255) & ~(size_t)255;
    return p;
  };
  _Float16* WqkvT = (_Float16*)alloc((size_t)3 * EE * EE * 2);   // [3072,1024]
  _Float16* WoT   = (_Float16*)alloc((size_t)EE * EE * 2);       // [1024,1024]
  _Float16* W1T   = (_Float16*)alloc((size_t)4 * EE * EE * 2);   // [4096,1024]
  _Float16* W2T   = (_Float16*)alloc((size_t)4 * EE * EE * 2);   // [1024,4096]
  _Float16* h1    = (_Float16*)alloc((size_t)MROWS * EE * 2);
  _Float16* qkv   = (_Float16*)alloc((size_t)3 * QKV_ONE * 2);
  _Float16* AO    = (_Float16*)alloc((size_t)MROWS * EE * 2);
  float*    x1    = (float*)   alloc((size_t)MROWS * EE * 4);
  _Float16* h2    = (_Float16*)alloc((size_t)MROWS * EE * 2);
  _Float16* ff    = (_Float16*)alloc((size_t)MROWS * 4 * EE * 2);

  // weight conversion (f32 -> f16 transposed)
  convert_qkv_w<<<(3 * EE * EE + 255) / 256, 256, 0, stream>>>(Wq, Wk, Wv, WqkvT);
  transpose_f16<<<(EE * EE + 255) / 256, 256, 0, stream>>>(Wo, WoT, EE, EE);
  transpose_f16<<<(4 * EE * EE + 255) / 256, 256, 0, stream>>>(W1, W1T, 4 * EE, EE);
  transpose_f16<<<(4 * EE * EE + 255) / 256, 256, 0, stream>>>(W2, W2T, EE, 4 * EE);

  // LN1
  ln_kernel<<<MROWS, 256, 0, stream>>>(x, ln1_g, ln1_b, h1);

  // fused QKV projection: [4096,1024] x [1024,3072]
  {
    dim3 g(MROWS / BM, 3072 / BN);
    wmma_gemm<0><<<g, 256, 0, stream>>>(h1, WqkvT, nullptr, nullptr,
                                        (void*)qkv, MROWS, 3072, EE);
  }

  // causal flash attention
  {
    dim3 g(SS / 64, BB * HH);
    attn_kernel<<<g, 128, 0, stream>>>(qkv, qkv + QKV_ONE,
                                       qkv + 2 * (size_t)QKV_ONE, AO);
  }

  // output projection + bias + residual -> x1 (fp32)
  {
    dim3 g(MROWS / BM, EE / BN);
    wmma_gemm<1><<<g, 256, 0, stream>>>(AO, WoT, bo, x, (void*)x1,
                                        MROWS, EE, EE);
  }

  // LN2
  ln_kernel<<<MROWS, 256, 0, stream>>>(x1, ln2_g, ln2_b, h2);

  // FFN1: relu(h2 @ W1 + b1) -> f16
  {
    dim3 g(MROWS / BM, (4 * EE) / BN);
    wmma_gemm<2><<<g, 256, 0, stream>>>(h2, W1T, b1, nullptr, (void*)ff,
                                        MROWS, 4 * EE, EE);
  }

  // FFN2: ff @ W2 + b2 + x1 -> out (fp32)
  {
    dim3 g(MROWS / BM, EE / BN);
    wmma_gemm<1><<<g, 256, 0, stream>>>(ff, W2T, b2, x1, (void*)out,
                                        MROWS, EE, 4 * EE);
  }
}
